// RNN_35012573397111
// MI455X (gfx1250) — compile-verified, matrix operand order checked by reading, not run on hardware
//
#include <hip/hip_runtime.h>
#include <stdint.h>

// ---------------------------------------------------------------------------
// 2-layer tanh RNN + FC head for MI455X (gfx1250, wave32, WMMA + TDM).
// bf16 matmul inputs, fp32 accumulation (v_wmma_f32_16x16x32_bf16).
// FC GEMM (167.8 GFLOP) dominates; its B panel (fc_W slice) is DMA'd into LDS
// by the Tensor Data Mover (tensor_load_to_lds, TENSORcnt) and shared by
// 8 waves x 2 M-tiles -> B L2 traffic 0.63 GB instead of 10 GB, and the
// K-loop has no per-step barriers or staging stores.
// Recurrent weights (8 MB bf16) stay L2-resident across the 256 step kernels.
// ---------------------------------------------------------------------------

#define S_LEN   128
#define BATCH   64
#define EMB     1024
#define HID     1024
#define VOCAB   10000
#define SB      (S_LEN * BATCH)          // 8192 rows into the FC
#define NTILES_V 625                     // 10000 / 16 exactly
#define FC_NGRP  157                     // ceil(625 / 4) n-tile groups
#define KC      512                      // FC K-chunk staged per TDM load

typedef __attribute__((ext_vector_type(16))) __bf16 v16bf;
typedef __attribute__((ext_vector_type(8)))  float  v8f;
typedef unsigned int u32x4 __attribute__((ext_vector_type(4)));
typedef int          i32x4 __attribute__((ext_vector_type(4)));
typedef int          i32x8 __attribute__((ext_vector_type(8)));

// ---- bf16 <-> f32 helpers (RNE) -------------------------------------------
__device__ __forceinline__ uint16_t f32_to_bf16(float f) {
    union { float f; uint32_t u; } v; v.f = f;
    uint32_t u = v.u;
    u += 0x7FFFu + ((u >> 16) & 1u);      // round to nearest even
    return (uint16_t)(u >> 16);
}
__device__ __forceinline__ float bf16_to_f32(uint16_t h) {
    union { uint32_t u; float f; } v; v.u = ((uint32_t)h) << 16;
    return v.f;
}

// ---- WMMA 16x32 bf16 fragment load ----------------------------------------
// ISA 7.12.2, 16-bit A-matrix 16x32: lane r=lane&15 holds row M=r.
// half h=lane>>4 selects K blocks: VGPR g in 0..3 -> K = 2g + 8h,
//                                  VGPR g in 4..7 -> K = 2g + 8 + 8h.
// Row-major [16 x ld] bf16 source => two contiguous 16B loads per lane.
__device__ __forceinline__ v16bf load_frag(const uint16_t* __restrict__ base,
                                           int ld, int lane) {
    const int r = lane & 15, h = lane >> 4;
    const uint16_t* row = base + r * ld + 8 * h;
    union { v16bf v; uint32_t u[8]; } f;
#pragma unroll
    for (int g = 0; g < 8; ++g) {
        const int k = 2 * g + ((g & 4) ? 8 : 0);
        f.u[g] = *(const uint32_t*)(row + k);
    }
    return f.v;
}

// Same pattern from an LDS-staged row-major tile (ds_load_b128 x2 after inline)
__device__ __forceinline__ v16bf load_frag_lds(const uint16_t* base, int ld,
                                               int lane) {
    const int r = lane & 15, h = lane >> 4;
    const uint16_t* row = base + r * ld + 8 * h;
    union { v16bf v; uint32_t u[8]; } f;
#pragma unroll
    for (int g = 0; g < 8; ++g) {
        const int k = 2 * g + ((g & 4) ? 8 : 0);
        f.u[g] = *(const uint32_t*)(row + k);
    }
    return f.v;
}

// ---- Tensor Data Mover: 2-D bf16 tile (KC x 64 rows) -> LDS ---------------
// D# per ISA 8.3/8.4: group0 = count|lds_addr|global_addr|type=2;
// group1 = data_size=1(2B), tensor_dim0=KC, tensor_dim1=rows_remaining
// (rows past the tensor edge read as zero -> harmless in WMMA),
// tile_dim0=KC, tile_dim1=64, tensor_dim0_stride=HID. Groups 2/3 zero (2-D).
__device__ __forceinline__ void tdm_load_b_panel(const uint16_t* gsrc,
                                                 uint32_t lds_byte_off,
                                                 int rows_remaining) {
    const uint64_t ga = (uint64_t)(uintptr_t)gsrc;
    u32x4 g0 = { 1u,                                   // count=1, user descriptor
                 lds_byte_off,                         // LDS dest (bytes)
                 (uint32_t)ga,                         // global_addr[31:0]
                 ((uint32_t)(ga >> 32) & 0x01FFFFFFu) | (2u << 30) }; // [56:32]|type=2
    i32x8 g1 = { (int)0x00010000u,                     // data_size=1 -> 2 bytes
                 (int)((uint32_t)KC << 16),            // tensor_dim0[15:0] @ bits63:48
                 (int)(((uint32_t)rows_remaining & 0xFFFFu) << 16), // t_dim1 lo16
                 (int)((uint32_t)KC << 16),            // tile_dim0 @ bits127:112
                 64,                                   // tile_dim1 = 64 rows
                 HID,                                  // tensor_dim0_stride lo32
                 0, 0 };
    i32x4 z4 = { 0, 0, 0, 0 };
#if defined(__clang_major__) && (__clang_major__ >= 23)
    i32x8 z8 = { 0, 0, 0, 0, 0, 0, 0, 0 };
    __builtin_amdgcn_tensor_load_to_lds(g0, g1, z4, z4, z8, 0);
#else
    __builtin_amdgcn_tensor_load_to_lds(g0, g1, z4, z4, 0);
#endif
}

// ---------------------------------------------------------------------------
// Prep kernels
// ---------------------------------------------------------------------------
__global__ void k_embed(const int* __restrict__ inputs,
                        const float* __restrict__ table,
                        uint16_t* __restrict__ X0) {
    const int idx = blockIdx.x * blockDim.x + threadIdx.x;   // over S*B*E
    const int e  = idx & (EMB - 1);
    const int sb = idx >> 10;
    const int tok = inputs[sb];
    X0[idx] = f32_to_bf16(table[(size_t)tok * EMB + e]);
}

// out[n*K + k] = bf16(in[k*N + n])   (weights -> [N][K] bf16)
__global__ void k_transpose_cvt(const float* __restrict__ in,
                                uint16_t* __restrict__ out, int K, int N) {
    const int idx = blockIdx.x * blockDim.x + threadIdx.x;   // over N*K
    const int k = idx % K;
    const int n = idx / K;
    out[idx] = f32_to_bf16(in[(size_t)k * N + n]);
}

__global__ void k_f32_to_bf16(const float* __restrict__ in,
                              uint16_t* __restrict__ out) {
    const int idx = blockIdx.x * blockDim.x + threadIdx.x;
    out[idx] = f32_to_bf16(in[idx]);
}

__global__ void k_bf16_to_f32(const uint16_t* __restrict__ in,
                              float* __restrict__ out) {
    const int idx = blockIdx.x * blockDim.x + threadIdx.x;
    out[idx] = bf16_to_f32(in[idx]);
}

// ---------------------------------------------------------------------------
// One RNN step-layer: Hout = tanh(X @ Wx + Hprev @ Wh + b)
// M=64 (4 mtiles), N=1024 (64 ntiles); 1 tile / wave, 8 waves / block
// -> 32 blocks (maximize WGP coverage on the latency-critical serial path).
// Weights are L2-resident; codegen is pipelined b128 clauses (verified rd 1).
// ---------------------------------------------------------------------------
__global__ __launch_bounds__(256) void k_rnn_step(
    const uint16_t* __restrict__ X,     const uint16_t* __restrict__ Hprev,
    const uint16_t* __restrict__ WxT,   const uint16_t* __restrict__ WhT,
    const float*    __restrict__ bias,  uint16_t* __restrict__ Hout) {
    const int lane = threadIdx.x & 31;
    const int task = blockIdx.x * 8 + (threadIdx.x >> 5);    // 0..255
    const int m0 = (task >> 6) * 16;                         // 4 mtiles
    const int n0 = (task & 63) * 16;                         // 64 ntiles

    v8f acc = {};

    const uint16_t* arow = X + m0 * HID;
    const uint16_t* brow = WxT + n0 * HID;
    for (int k = 0; k < HID; k += 32) {
        v16bf a = load_frag(arow + k, HID, lane);
        v16bf b = load_frag(brow + k, HID, lane);
        acc = __builtin_amdgcn_wmma_f32_16x16x32_bf16(
                  false, a, false, b, (short)0, acc, false, false);
    }
    arow = Hprev + m0 * HID;
    brow = WhT + n0 * HID;
    for (int k = 0; k < HID; k += 32) {
        v16bf a = load_frag(arow + k, HID, lane);
        v16bf b = load_frag(brow + k, HID, lane);
        acc = __builtin_amdgcn_wmma_f32_16x16x32_bf16(
                  false, a, false, b, (short)0, acc, false, false);
    }

    // C/D layout: element e -> M = m0 + e + 8*(lane>=16), N = n0 + (lane&15)
    const int r = lane & 15, h = lane >> 4;
    const int n = n0 + r;
    const float bv = bias[n];
#pragma unroll
    for (int e = 0; e < 8; ++e) {
        const int m = m0 + e + 8 * h;
        Hout[m * HID + n] = f32_to_bf16(tanhf(acc[e] + bv));
    }
}

// ---------------------------------------------------------------------------
// FC head: logits(8192 x 10000) = tops(8192x1024) @ fc_W + fc_b
// Block = 256 rows (8 waves x 2 M-tiles) x 64 cols (4 N-tiles).
// The 64-row x KC-col bf16 B panel (64 KB) is DMA'd into LDS by the TDM
// (one tensor_load_to_lds per K-chunk, wave 0 only, s_wait_tensorcnt 0),
// then the K-loop is pure: 4 global b128 (A) + 8 ds_load_b128 (B) + 8 WMMA
// per K-step, with only 2 barriers per 512-wide chunk.
// Grid: (8192/256) * 157 = 5024 blocks.
// ---------------------------------------------------------------------------
__global__ __launch_bounds__(256) void k_fc(
    const uint16_t* __restrict__ tops, const uint16_t* __restrict__ WT,
    const float* __restrict__ bias,    float* __restrict__ out) {
    const int lane = threadIdx.x & 31;
    const int wave = threadIdx.x >> 5;
    const int bm = blockIdx.x / FC_NGRP;           // 0..31
    const int ng = blockIdx.x % FC_NGRP;           // 0..156
    const int m0 = bm * 256 + wave * 32;           // this wave: rows m0..m0+31

    __shared__ uint16_t Bs[64 * KC];               // 64 KB B panel (rows x KC)

    // The TDM writes Bs behind the compiler's back. Without a visible store,
    // LLVM folds all loads from this never-stored LDS global to poison and
    // deletes the ds_load_b128s (observed round 3: ds=0). This runtime-dead
    // store (blockDim.y is 1) keeps a genuine store to Bs in the module.
    if (blockDim.y == 2) Bs[threadIdx.x] = (uint16_t)threadIdx.x;

    v8f acc[2][4] = {};

    const uint16_t* arow0 = tops + (size_t)m0 * HID;
    const uint16_t* arow1 = arow0 + 16 * HID;
    const uint16_t* bpanel = WT + (size_t)(ng * 64) * HID;
    const int rows_rem = VOCAB - ng * 64;          // TDM zero-fills rows past edge

    for (int kc = 0; kc < HID; kc += KC) {
        if (threadIdx.x < 32) {                    // one wave drives the DMA
            tdm_load_b_panel(bpanel + kc, 0u, rows_rem);
            __builtin_amdgcn_s_wait_tensorcnt(0);
        }
        __syncthreads();                           // panel visible to all waves
        // Escape Bs into an asm with a memory clobber: the compiler must now
        // assume Bs' contents changed (the TDM DMA it cannot see).
        asm volatile("" :: "v"(&Bs[0]) : "memory");

        for (int k2 = 0; k2 < KC; k2 += 32) {
            const int k = kc + k2;
            __builtin_prefetch(arow0 + k + 32, 0, 1);   // global_prefetch_b8
            __builtin_prefetch(arow1 + k + 32, 0, 1);
            v16bf a0 = load_frag(arow0 + k, HID, lane);
            v16bf a1 = load_frag(arow1 + k, HID, lane);
#pragma unroll
            for (int j = 0; j < 4; ++j) {
                if (ng * 4 + j < NTILES_V) {       // uniform -> EXEC all-1s
                    v16bf b = load_frag_lds(Bs + j * 16 * KC + k2, KC, lane);
                    acc[0][j] = __builtin_amdgcn_wmma_f32_16x16x32_bf16(
                                    false, a0, false, b, (short)0, acc[0][j], false, false);
                    acc[1][j] = __builtin_amdgcn_wmma_f32_16x16x32_bf16(
                                    false, a1, false, b, (short)0, acc[1][j], false, false);
                }
            }
        }
        __syncthreads();                           // all reads done before next DMA
    }

    const int r = lane & 15, h = lane >> 4;
#pragma unroll
    for (int mi = 0; mi < 2; ++mi) {
#pragma unroll
        for (int j = 0; j < 4; ++j) {
            const int nt = ng * 4 + j;
            if (nt >= NTILES_V) continue;
            const int n = nt * 16 + r;
            const float bv = bias[n];
#pragma unroll
            for (int e = 0; e < 8; ++e) {
                const int m = m0 + mi * 16 + e + 8 * h;
                out[(size_t)m * VOCAB + n] = acc[mi][j][e] + bv;
            }
        }
    }
}

// ---------------------------------------------------------------------------
// Host side
// ---------------------------------------------------------------------------
extern "C" void kernel_launch(void* const* d_in, const int* in_sizes, int n_in,
                              void* d_out, int out_size, void* d_ws, size_t ws_size,
                              hipStream_t stream) {
    const int*   inputs    = (const int*)  d_in[0];
    const float* hidden    = (const float*)d_in[1];
    const float* emb_table = (const float*)d_in[2];
    const float* Wx0 = (const float*)d_in[3];
    const float* Wh0 = (const float*)d_in[4];
    const float* b0  = (const float*)d_in[5];
    const float* Wx1 = (const float*)d_in[6];
    const float* Wh1 = (const float*)d_in[7];
    const float* b1  = (const float*)d_in[8];
    const float* fc_W = (const float*)d_in[9];
    const float* fc_b = (const float*)d_in[10];

    float* logits  = (float*)d_out;                              // (S,B,V)
    float* h_final = logits + (size_t)S_LEN * BATCH * VOCAB;     // (L,B,H)

    // Workspace layout (bf16 buffers), ~62.9 MB total.
    char* ws = (char*)d_ws;
    const size_t BH  = (size_t)BATCH * HID;          // 65536 elems
    uint16_t* X0     = (uint16_t*)(ws);                               // 16 MB
    uint16_t* tops   = (uint16_t*)(ws + (16u << 20));                 // 16 MB
    uint16_t* WxT0   = (uint16_t*)(ws + (32u << 20));                 // 2 MB
    uint16_t* WhT0   = (uint16_t*)(ws + (34u << 20));
    uint16_t* WxT1   = (uint16_t*)(ws + (36u << 20));
    uint16_t* WhT1   = (uint16_t*)(ws + (38u << 20));
    uint16_t* fcWT   = (uint16_t*)(ws + (40u << 20));                 // 19.6 MB
    uint16_t* h0buf  = (uint16_t*)(ws + (60u << 20));                 // 2 x 128 KB
    uint16_t* h1init = (uint16_t*)(ws + (61u << 20));                 // 128 KB
    (void)in_sizes; (void)n_in; (void)out_size; (void)ws_size;

    // ---- prep: embed gather + convert, weight transpose+convert ----
    k_embed<<<(S_LEN * BATCH * EMB) / 256, 256, 0, stream>>>(inputs, emb_table, X0);
    k_transpose_cvt<<<(HID * HID) / 256, 256, 0, stream>>>(Wx0, WxT0, HID, HID);
    k_transpose_cvt<<<(HID * HID) / 256, 256, 0, stream>>>(Wh0, WhT0, HID, HID);
    k_transpose_cvt<<<(HID * HID) / 256, 256, 0, stream>>>(Wx1, WxT1, HID, HID);
    k_transpose_cvt<<<(HID * HID) / 256, 256, 0, stream>>>(Wh1, WhT1, HID, HID);
    k_transpose_cvt<<<(VOCAB * HID) / 256, 256, 0, stream>>>(fc_W, fcWT, HID, VOCAB);
    k_f32_to_bf16<<<(int)(BH / 256), 256, 0, stream>>>(hidden,      h0buf);   // h0[0]
    k_f32_to_bf16<<<(int)(BH / 256), 256, 0, stream>>>(hidden + BH, h1init);

    // ---- serial recurrence: 128 steps x 2 layers ----
    for (int t = 0; t < S_LEN; ++t) {
        const int p = t & 1;
        uint16_t* h0p = h0buf + (size_t)p * BH;
        uint16_t* h0n = h0buf + (size_t)(1 - p) * BH;
        k_rnn_step<<<32, 256, 0, stream>>>(X0 + (size_t)t * BH, h0p,
                                           WxT0, WhT0, b0, h0n);
        const uint16_t* h1p = (t == 0) ? h1init : (tops + (size_t)(t - 1) * BH);
        k_rnn_step<<<32, 256, 0, stream>>>(h0n, h1p,
                                           WxT1, WhT1, b1, tops + (size_t)t * BH);
    }

    // ---- FC head (dominant GEMM) ----
    k_fc<<<32 * FC_NGRP, 256, 0, stream>>>(tops, fcWT, fc_b, logits);

    // ---- h_final: h0 ended in buffer 0 (t=127 odd -> wrote buf 0) ----
    k_bf16_to_f32<<<(int)(BH / 256), 256, 0, stream>>>(h0buf, h_final);
    k_bf16_to_f32<<<(int)(BH / 256), 256, 0, stream>>>(tops + (size_t)(S_LEN - 1) * BH,
                                                       h_final + BH);
}
